// EnvironmentConditionalTransformer_16355235463523
// MI455X (gfx1250) — compile-verified, matrix operand order checked by reading, not run on hardware
//
#include <hip/hip_runtime.h>
#include <hip/hip_bf16.h>

// ---------------- problem constants (match reference) ----------------
constexpr int NB = 4;      // batch
constexpr int NS = 4096;   // SNPs
constexpr int NG = 1024;   // genes
constexpr int NP = 20;     // PCs
constexpr int ND = 128;    // model dim
constexpr int NH = 4;      // heads
constexpr int NDK = 32;    // head dim
constexpr int NL = 2;      // layers
constexpr float EPSV = 1e-5f;
constexpr float NEGV = -1e9f;

typedef __attribute__((ext_vector_type(16))) _Float16 v16h;
typedef __attribute__((ext_vector_type(8)))  _Float16 v8h;
typedef __attribute__((ext_vector_type(8)))  float    v8f;
typedef __attribute__((ext_vector_type(4)))  unsigned u32x4;
typedef __attribute__((ext_vector_type(8)))  unsigned u32x8;

// ---------------- wave32 helpers ----------------
__device__ __forceinline__ float wred_sum32(float v) {
#pragma unroll
  for (int m = 16; m >= 1; m >>= 1) v += __shfl_xor(v, m, 32);
  return v;
}
__device__ __forceinline__ float wred_sum16(float v) {
#pragma unroll
  for (int m = 8; m >= 1; m >>= 1) v += __shfl_xor(v, m, 32);
  return v;
}
__device__ __forceinline__ float wred_max16(float v) {
#pragma unroll
  for (int m = 8; m >= 1; m >>= 1) v = fmaxf(v, __shfl_xor(v, m, 32));
  return v;
}

// wave-local LDS RAW fence (writes by lanes -> reads by other lanes, same wave)
__device__ __forceinline__ void lds_fence() {
  __builtin_amdgcn_wave_barrier();
#if __has_builtin(__builtin_amdgcn_s_wait_dscnt)
  __builtin_amdgcn_s_wait_dscnt(0);
#else
  asm volatile("s_wait_dscnt 0" ::: "memory");
#endif
  __builtin_amdgcn_wave_barrier();
}

// ---------------- CDNA5 async global->LDS staging ----------------
// 32-bit LDS byte address of a __shared__ object (generic -> AS(3) cast).
__device__ __forceinline__ unsigned lds_lo(const void* p) {
  return (unsigned)(unsigned long long)(uintptr_t)
         (const __attribute__((address_space(3))) char*)p;
}
// GLOBAL_LOAD_ASYNC_TO_LDS_B128: per-lane 16B memory -> LDS, tracked by ASYNCcnt.
__device__ __forceinline__ void async_b128(unsigned lds, const _Float16* g) {
  asm volatile("global_load_async_to_lds_b128 %0, %1, off"
               :: "v"(lds), "v"((unsigned long long)(uintptr_t)g)
               : "memory");
}
__device__ __forceinline__ void wait_async_le4() {
#if __has_builtin(__builtin_amdgcn_s_wait_asynccnt)
  __builtin_amdgcn_s_wait_asynccnt(4);
#else
  asm volatile("s_wait_asynccnt 0x4" ::: "memory");
#endif
}
__device__ __forceinline__ void wait_async_le0() {
#if __has_builtin(__builtin_amdgcn_s_wait_asynccnt)
  __builtin_amdgcn_s_wait_asynccnt(0);
#else
  asm volatile("s_wait_asynccnt 0x0" ::: "memory");
#endif
}

// ---------------- CDNA5 Tensor Data Mover: 2D tile global -> LDS ----------------
// Loads a tile of 32 rows x 32 f16 (64B/row, row stride = 128 f16) into a compact
// 32x32 LDS buffer. D# groups 0/1 built from wave-uniform scalars; VADDR2/3 = NULL
// (<=2D tensor). Tracked by TENSORcnt.
__device__ __forceinline__ void tdm_load_k32x32(unsigned lds, const _Float16* g) {
  unsigned long long ga = (unsigned long long)(uintptr_t)g;
  unsigned lds_u  = (unsigned)__builtin_amdgcn_readfirstlane((int)lds);
  unsigned ga_lo  = (unsigned)__builtin_amdgcn_readfirstlane((int)(unsigned)ga);
  unsigned ga_hi  = (unsigned)__builtin_amdgcn_readfirstlane((int)(unsigned)(ga >> 32));
  u32x4 g0;
  g0[0] = 1u;                          // count=1 (valid user descriptor)
  g0[1] = lds_u;                       // lds_addr (bytes)
  g0[2] = ga_lo;                       // global_addr[31:0]
  g0[3] = (ga_hi & 0x01ffffffu) | (2u << 30);  // global_addr[56:32] | type=2 (image)
  u32x8 g1;
  g1[0] = 1u << 16;                    // workgroup_mask=0, data_size=1 (2 bytes)
  g1[1] = 32u << 16;                   // tensor_dim0 = 32 elems
  g1[2] = 32u << 16;                   // tensor_dim1 = 32 rows
  g1[3] = 32u << 16;                   // tile_dim0 = 32 elems
  g1[4] = 32u;                         // tile_dim1 = 32 rows (tile_dim2 = 0)
  g1[5] = (unsigned)ND;                // tensor_dim0_stride = 128 elems
  g1[6] = 0u;
  g1[7] = 0u;
  asm volatile("tensor_load_to_lds %0, %1" :: "s"(g0), "s"(g1) : "memory");
}
__device__ __forceinline__ void wait_tensor_le1() {
#if __has_builtin(__builtin_amdgcn_s_wait_tensorcnt)
  __builtin_amdgcn_s_wait_tensorcnt(1);
#else
  asm volatile("s_wait_tensorcnt 0x1" ::: "memory");
#endif
}
__device__ __forceinline__ void wait_tensor_le0() {
#if __has_builtin(__builtin_amdgcn_s_wait_tensorcnt)
  __builtin_amdgcn_s_wait_tensorcnt(0);
#else
  asm volatile("s_wait_tensorcnt 0x0" ::: "memory");
#endif
}

// ---------------- env prep: dq/dk + FiLM gamma/beta ----------------
__global__ void env_prep(const float* __restrict__ env_emb, const int* __restrict__ env_idx,
                         const float* __restrict__ wq_env, const float* __restrict__ wk_env,
                         const float* __restrict__ gamma_w, const float* __restrict__ gamma_b,
                         const float* __restrict__ beta_w, const float* __restrict__ beta_b,
                         float* __restrict__ dq, float* __restrict__ dk,
                         float* __restrict__ gam, float* __restrict__ bet) {
  int t = blockIdx.x * blockDim.x + threadIdx.x;
  if (t >= NB * ND) return;
  int bb = t / ND, d = t % ND;
  const float* e = env_emb + (size_t)env_idx[bb] * ND;
  float sq = 0.f, sk = 0.f, sg = 0.f, sb = 0.f;
  for (int k = 0; k < ND; ++k) {
    float ev = e[k];
    sq += ev * wq_env[k * ND + d];
    sk += ev * wk_env[k * ND + d];
    sg += ev * gamma_w[k * ND + d];
    sb += ev * beta_w[k * ND + d];
  }
  dq[t] = sq;
  dk[t] = sk;
  gam[t] = tanhf(sg + gamma_b[d]);
  bet[t] = sb + beta_b[d];
}

// ---------------- SNP encoder: wave per (b,s) row ----------------
__global__ void snp_encoder(const float* __restrict__ genotypes, const float* __restrict__ pcs,
                            const float* __restrict__ w1, const float* __restrict__ b1,
                            const float* __restrict__ g1, const float* __restrict__ be1,
                            const float* __restrict__ w2, const float* __restrict__ b2,
                            const float* __restrict__ g2, const float* __restrict__ be2,
                            _Float16* __restrict__ snpF) {
  __shared__ float Hl[8][64];
  int w = threadIdx.x >> 5, lane = threadIdx.x & 31;
  int row = blockIdx.x * 8 + w;
  if (row >= NB * NS) return;
  int bb = row / NS;
  float geno = genotypes[row];
  const float* pc = pcs + bb * NP;
  float h[2];
#pragma unroll
  for (int t = 0; t < 2; ++t) {
    int j = lane + t * 32;
    float acc = b1[j] + geno * w1[j];
    for (int k = 0; k < NP; ++k) acc += pc[k] * w1[(1 + k) * 64 + j];
    h[t] = acc;
  }
  float mean = wred_sum32(h[0] + h[1]) * (1.f / 64.f);
  float d0 = h[0] - mean, d1 = h[1] - mean;
  float var = wred_sum32(d0 * d0 + d1 * d1) * (1.f / 64.f);
  float rstd = rsqrtf(var + EPSV);
#pragma unroll
  for (int t = 0; t < 2; ++t) {
    int j = lane + t * 32;
    float y = (h[t] - mean) * rstd * g1[j] + be1[j];
    Hl[w][j] = fmaxf(y, 0.f);
  }
  lds_fence();
  float o[4];
#pragma unroll
  for (int c = 0; c < 4; ++c) {
    int j = lane + c * 32;
    float acc = b2[j];
    for (int k = 0; k < 64; ++k) acc += Hl[w][k] * w2[k * ND + j];
    o[c] = acc;
  }
  float mean2 = wred_sum32(o[0] + o[1] + o[2] + o[3]) * (1.f / 128.f);
  float vs = 0.f;
#pragma unroll
  for (int c = 0; c < 4; ++c) { float dd = o[c] - mean2; vs += dd * dd; }
  float rstd2 = rsqrtf(wred_sum32(vs) * (1.f / 128.f) + EPSV);
#pragma unroll
  for (int c = 0; c < 4; ++c) {
    int j = lane + c * 32;
    snpF[(size_t)row * ND + j] = (_Float16)((o[c] - mean2) * rstd2 * g2[j] + be2[j]);
  }
}

// ---------------- weight f32 -> f16 conversion ----------------
__global__ void conv_w(const float* __restrict__ wq, const float* __restrict__ wk,
                       const float* __restrict__ wv, const float* __restrict__ wo,
                       _Float16* __restrict__ w16) {
  int i = blockIdx.x * blockDim.x + threadIdx.x;
  const int tot = NL * ND * ND;
  if (i >= tot) return;
  w16[i] = (_Float16)wq[i];
  w16[tot + i] = (_Float16)wk[i];
  w16[2 * tot + i] = (_Float16)wv[i];
  w16[3 * tot + i] = (_Float16)wo[i];
}

// ---------------- hidden init from gene embedding ----------------
__global__ void hid_init(const float* __restrict__ gene_emb, float* __restrict__ hid32,
                         _Float16* __restrict__ hid16) {
  int i = blockIdx.x * blockDim.x + threadIdx.x;
  if (i >= NB * NG * ND) return;
  float v = gene_emb[i % (NG * ND)];
  hid32[i] = v;
  hid16[i] = (_Float16)v;
}

// ---------------- generic WMMA GEMM: out[b] = A[b] @ W + bias + extra[b] ----------------
template <int OUTF32>
__global__ void gemm_wmma(const _Float16* __restrict__ A, const _Float16* __restrict__ W,
                          const float* __restrict__ bias, const float* __restrict__ extra,
                          _Float16* __restrict__ outH, float* __restrict__ outF,
                          int M, int N, int Kd, int nbatch) {
  int wid = blockIdx.x * (blockDim.x >> 5) + (threadIdx.x >> 5);
  int lane = threadIdx.x & 31;
  int ntiles = N >> 4, mtiles = M >> 4;
  if (wid >= nbatch * mtiles * ntiles) return;   // wave-uniform
  int nt = wid % ntiles;
  int mt = (wid / ntiles) % mtiles;
  int bb = wid / (ntiles * mtiles);
  int rr = lane & 15, half = lane >> 4;
  int n = nt * 16 + rr;
  const _Float16* Ab = A + (size_t)bb * M * Kd + (size_t)(mt * 16) * Kd;
  v8f c = {};
  for (int kk = 0; kk < Kd; kk += 32) {
    v16h a;
    *(v8h*)&a       = *(const v8h*)(Ab + (size_t)rr * Kd + kk + half * 8);
    *((v8h*)&a + 1) = *(const v8h*)(Ab + (size_t)rr * Kd + kk + 16 + half * 8);
    v16h b;
#pragma unroll
    for (int j = 0; j < 8; ++j) {
      b[j]     = W[(size_t)(kk + half * 8 + j) * N + n];
      b[8 + j] = W[(size_t)(kk + 16 + half * 8 + j) * N + n];
    }
    c = __builtin_amdgcn_wmma_f32_16x16x32_f16(false, a, false, b, (short)0, c, false, false);
  }
  float add = (bias ? bias[n] : 0.f) + (extra ? extra[bb * N + n] : 0.f);
  size_t obase = (size_t)bb * M * N + (size_t)(mt * 16) * N + n;
#pragma unroll
  for (int r = 0; r < 8; ++r) {
    int m = half * 8 + r;
    float val = c[r] + add;
    if (OUTF32) outF[obase + (size_t)m * N] = val;
    else        outH[obase + (size_t)m * N] = (_Float16)val;
  }
}

// ---------------- WMMA flash attention, double-buffered K via TDM, V via async ----------------
__global__ void attn_flash(const _Float16* __restrict__ Q, const _Float16* __restrict__ Kb,
                           const _Float16* __restrict__ Vb, const unsigned char* __restrict__ cmask,
                           _Float16* __restrict__ att) {
  __shared__ __attribute__((aligned(32))) _Float16 KtS[8][2][32][32];
  __shared__ __attribute__((aligned(32))) _Float16 VtS[8][2][32][32];
  __shared__ __attribute__((aligned(32))) _Float16 Pl[8][16][32];
  int w = threadIdx.x >> 5, lane = threadIdx.x & 31;
  int wid = blockIdx.x * 8 + w;
  const int GT = NG / 16;
  if (wid >= NB * NH * GT) return;   // wave-uniform
  int gt = wid % GT;
  int h = (wid / GT) % NH;
  int bb = wid / (GT * NH);
  int rr = lane & 15, half = lane >> 4;

  const _Float16* Qb = Q + (size_t)bb * NG * ND + (size_t)(gt * 16) * ND + h * NDK;
  const _Float16* Kh = Kb + (size_t)bb * NS * ND + h * NDK;
  const _Float16* Vh = Vb + (size_t)bb * NS * ND + h * NDK;

  // stage one 32-SNP chunk: K tile via TDM (1 op), V rows via async b128 (4 ops)
  auto stage = [&](int buf, int s0n) {
    tdm_load_k32x32(lds_lo(&KtS[w][buf][0][0]), Kh + (size_t)s0n * ND);
    const _Float16* vg = Vh + (size_t)(s0n + lane) * ND;
    unsigned vl = lds_lo(&VtS[w][buf][lane][0]);
#pragma unroll
    for (int t = 0; t < 4; ++t) {
      async_b128(vl + t * 16, vg + t * 8);
    }
  };

  v16h aq;
  *(v8h*)&aq       = *(const v8h*)(Qb + (size_t)rr * ND + half * 8);
  *((v8h*)&aq + 1) = *(const v8h*)(Qb + (size_t)rr * ND + 16 + half * 8);

  float m[8], l[8];
  v8f o0 = {}, o1 = {};
#pragma unroll
  for (int r = 0; r < 8; ++r) { m[r] = -3.0e38f; l[r] = 0.f; }
  const float scale = 0.17677669529663687f;  // 1/sqrt(32)

  const int nchunks = NS / 32;
  stage(0, 0);
  for (int ci = 0; ci < nchunks; ++ci) {
    int s0 = ci * 32;
    int cur = ci & 1;
    bool more = (ci + 1) < nchunks;
    if (more) stage(cur ^ 1, s0 + 32);  // prefetch next chunk on both DMA engines
    if (more) { wait_tensor_le1(); wait_async_le4(); }
    else      { wait_tensor_le0(); wait_async_le0(); }
    __builtin_amdgcn_wave_barrier();

    const _Float16(*kt)[32] = KtS[w][cur];
    const _Float16(*vt)[32] = VtS[w][cur];

    // B-fragments of K^T for two 16-SNP sub-tiles (contiguous LDS reads)
    v16h b0, b1;
    *(v8h*)&b0       = *(const v8h*)&kt[rr][half * 8];
    *((v8h*)&b0 + 1) = *(const v8h*)&kt[rr][16 + half * 8];
    *(v8h*)&b1       = *(const v8h*)&kt[16 + rr][half * 8];
    *((v8h*)&b1 + 1) = *(const v8h*)&kt[16 + rr][16 + half * 8];

    v8f z = {};
    v8f sc0 = __builtin_amdgcn_wmma_f32_16x16x32_f16(false, aq, false, b0, (short)0, z, false, false);
    v8f sc1 = __builtin_amdgcn_wmma_f32_16x16x32_f16(false, aq, false, b1, (short)0, z, false, false);

    // mask, scale, online softmax; stash P (f16) to LDS in C-layout
#pragma unroll
    for (int r = 0; r < 8; ++r) {
      int g = gt * 16 + half * 8 + r;
      int sA = s0 + rr, sB = s0 + 16 + rr;
      float v0 = cmask[(size_t)g * NS + sA] ? sc0[r] * scale : NEGV;
      float v1 = cmask[(size_t)g * NS + sB] ? sc1[r] * scale : NEGV;
      float cm = wred_max16(fmaxf(v0, v1));
      float mn = fmaxf(m[r], cm);
      float alpha = __expf(m[r] - mn);
      float p0 = __expf(v0 - mn), p1 = __expf(v1 - mn);
      float rs = wred_sum16(p0 + p1);
      l[r] = l[r] * alpha + rs;
      m[r] = mn;
      o0[r] *= alpha; o1[r] *= alpha;
      Pl[w][half * 8 + r][rr] = (_Float16)p0;
      Pl[w][half * 8 + r][16 + rr] = (_Float16)p1;
    }
    lds_fence();
    // reload P in A-layout (16 genes x 32 SNPs)
    v16h ap;
    *(v8h*)&ap       = *(const v8h*)&Pl[w][rr][half * 8];
    *((v8h*)&ap + 1) = *(const v8h*)&Pl[w][rr][16 + half * 8];
    // B-fragments of V chunk (32 SNPs x 32 dk) gathered from LDS
    v16h bv0, bv1;
#pragma unroll
    for (int j = 0; j < 8; ++j) {
      bv0[j]     = vt[half * 8 + j][rr];
      bv0[8 + j] = vt[16 + half * 8 + j][rr];
      bv1[j]     = vt[half * 8 + j][16 + rr];
      bv1[8 + j] = vt[16 + half * 8 + j][16 + rr];
    }
    o0 = __builtin_amdgcn_wmma_f32_16x16x32_f16(false, ap, false, bv0, (short)0, o0, false, false);
    o1 = __builtin_amdgcn_wmma_f32_16x16x32_f16(false, ap, false, bv1, (short)0, o1, false, false);
    lds_fence();  // protect Pl before next iteration overwrites it
  }

  _Float16* ob = att + (size_t)bb * NG * ND + (size_t)(gt * 16) * ND + h * NDK;
#pragma unroll
  for (int r = 0; r < 8; ++r) {
    float inv = l[r] > 0.f ? 1.f / l[r] : 0.f;
    int mrow = half * 8 + r;
    ob[(size_t)mrow * ND + rr]      = (_Float16)(o0[r] * inv);
    ob[(size_t)mrow * ND + 16 + rr] = (_Float16)(o1[r] * inv);
  }
}

// ---------------- LN + residual: wave per (b,g) row ----------------
__global__ void ln_residual(const float* __restrict__ tmp, const float* __restrict__ lng,
                            const float* __restrict__ lnb, float* __restrict__ hid32,
                            _Float16* __restrict__ hid16) {
  int w = threadIdx.x >> 5, lane = threadIdx.x & 31;
  int row = blockIdx.x * 8 + w;
  if (row >= NB * NG) return;
  float x[4];
#pragma unroll
  for (int c = 0; c < 4; ++c) x[c] = tmp[(size_t)row * ND + lane + c * 32];
  float mean = wred_sum32(x[0] + x[1] + x[2] + x[3]) * (1.f / 128.f);
  float vs = 0.f;
#pragma unroll
  for (int c = 0; c < 4; ++c) { float dd = x[c] - mean; vs += dd * dd; }
  float rstd = rsqrtf(wred_sum32(vs) * (1.f / 128.f) + EPSV);
#pragma unroll
  for (int c = 0; c < 4; ++c) {
    int j = lane + c * 32;
    float y = (x[c] - mean) * rstd * lng[j] + lnb[j];
    float nh = hid32[(size_t)row * ND + j] + y;
    hid32[(size_t)row * ND + j] = nh;
    hid16[(size_t)row * ND + j] = (_Float16)nh;
  }
}

// ---------------- FiLM + prediction head: wave per (b,g) row ----------------
__global__ void film_head(const float* __restrict__ hid32, const float* __restrict__ gam,
                          const float* __restrict__ bet, const float* __restrict__ hw1,
                          const float* __restrict__ hb1, const float* __restrict__ hg,
                          const float* __restrict__ hbb, const float* __restrict__ hw2,
                          const float* __restrict__ hb2, float* __restrict__ out) {
  __shared__ float Rl[8][128];
  int w = threadIdx.x >> 5, lane = threadIdx.x & 31;
  int row = blockIdx.x * 8 + w;
  if (row >= NB * NG) return;
  int bb = row / NG;
#pragma unroll
  for (int c = 0; c < 4; ++c) {
    int j = lane + c * 32;
    float rep = hid32[(size_t)row * ND + j] * (1.f + gam[bb * ND + j]) + bet[bb * ND + j];
    Rl[w][j] = rep;
  }
  lds_fence();
  float h[2];
#pragma unroll
  for (int t = 0; t < 2; ++t) {
    int j = lane + t * 32;
    float acc = hb1[j];
    for (int k = 0; k < 128; ++k) acc += Rl[w][k] * hw1[k * 64 + j];
    h[t] = acc;
  }
  float mean = wred_sum32(h[0] + h[1]) * (1.f / 64.f);
  float d0 = h[0] - mean, d1 = h[1] - mean;
  float rstd = rsqrtf(wred_sum32(d0 * d0 + d1 * d1) * (1.f / 64.f) + EPSV);
  float y0 = fmaxf((h[0] - mean) * rstd * hg[lane] + hbb[lane], 0.f);
  float y1 = fmaxf((h[1] - mean) * rstd * hg[lane + 32] + hbb[lane + 32], 0.f);
  float part = wred_sum32(y0 * hw2[lane] + y1 * hw2[lane + 32]);
  if (lane == 0) out[row] = part + hb2[0];
}

// ---------------- host orchestration ----------------
extern "C" void kernel_launch(void* const* d_in, const int* in_sizes, int n_in,
                              void* d_out, int out_size, void* d_ws, size_t ws_size,
                              hipStream_t stream) {
  (void)in_sizes; (void)n_in; (void)out_size; (void)ws_size;
  const float* genotypes = (const float*)d_in[0];
  const float* pcs       = (const float*)d_in[1];
  const int*   env_idx   = (const int*)d_in[2];
  const unsigned char* cmask = (const unsigned char*)d_in[3];
  const float* snp_w1 = (const float*)d_in[4];
  const float* snp_b1 = (const float*)d_in[5];
  const float* snp_g1 = (const float*)d_in[6];
  const float* snp_be1 = (const float*)d_in[7];
  const float* snp_w2 = (const float*)d_in[8];
  const float* snp_b2 = (const float*)d_in[9];
  const float* snp_g2 = (const float*)d_in[10];
  const float* snp_be2 = (const float*)d_in[11];
  const float* env_emb = (const float*)d_in[12];
  const float* wq_env = (const float*)d_in[13];
  const float* wk_env = (const float*)d_in[14];
  const float* gamma_w = (const float*)d_in[15];
  const float* gamma_b = (const float*)d_in[16];
  const float* beta_w = (const float*)d_in[17];
  const float* beta_b = (const float*)d_in[18];
  const float* gene_emb = (const float*)d_in[19];
  const float* wq = (const float*)d_in[20];
  const float* bq = (const float*)d_in[21];
  const float* wk = (const float*)d_in[22];
  const float* bk = (const float*)d_in[23];
  const float* wv = (const float*)d_in[24];
  const float* bv = (const float*)d_in[25];
  const float* wo = (const float*)d_in[26];
  const float* bo = (const float*)d_in[27];
  const float* lng = (const float*)d_in[28];
  const float* lnb = (const float*)d_in[29];
  const float* hw1 = (const float*)d_in[30];
  const float* hb1 = (const float*)d_in[31];
  const float* hlg = (const float*)d_in[32];
  const float* hlb = (const float*)d_in[33];
  const float* hw2 = (const float*)d_in[34];
  const float* hb2 = (const float*)d_in[35];
  float* out = (float*)d_out;

  // ---- workspace carve (all 256B aligned) ----
  char* wsb = (char*)d_ws;
  size_t off = 0;
  auto carve = [&](size_t bytes) -> char* {
    char* p = wsb + off;
    off += (bytes + 255) & ~((size_t)255);
    return p;
  };
  _Float16* snpF  = (_Float16*)carve((size_t)NB * NS * ND * 2);
  _Float16* K16   = (_Float16*)carve((size_t)NB * NS * ND * 2);
  _Float16* V16   = (_Float16*)carve((size_t)NB * NS * ND * 2);
  _Float16* Q16   = (_Float16*)carve((size_t)NB * NG * ND * 2);
  _Float16* att16 = (_Float16*)carve((size_t)NB * NG * ND * 2);
  float*    hid32 = (float*)carve((size_t)NB * NG * ND * 4);
  _Float16* hid16 = (_Float16*)carve((size_t)NB * NG * ND * 2);
  float*    tmp32 = (float*)carve((size_t)NB * NG * ND * 4);
  _Float16* w16   = (_Float16*)carve((size_t)4 * NL * ND * ND * 2);
  float* dq  = (float*)carve((size_t)NB * ND * 4);
  float* dkv = (float*)carve((size_t)NB * ND * 4);
  float* gam = (float*)carve((size_t)NB * ND * 4);
  float* bet = (float*)carve((size_t)NB * ND * 4);

  const int WMAT = NL * ND * ND;  // elems per weight stack

  env_prep<<<(NB * ND + 255) / 256, 256, 0, stream>>>(
      env_emb, env_idx, wq_env, wk_env, gamma_w, gamma_b, beta_w, beta_b, dq, dkv, gam, bet);

  snp_encoder<<<(NB * NS) / 8, 256, 0, stream>>>(
      genotypes, pcs, snp_w1, snp_b1, snp_g1, snp_be1, snp_w2, snp_b2, snp_g2, snp_be2, snpF);

  conv_w<<<(WMAT + 255) / 256, 256, 0, stream>>>(wq, wk, wv, wo, w16);

  hid_init<<<(NB * NG * ND + 255) / 256, 256, 0, stream>>>(gene_emb, hid32, hid16);

  const int gemmG_blocks = NB * (NG / 16) * (ND / 16) / 8;  // 256
  const int gemmS_blocks = NB * (NS / 16) * (ND / 16) / 8;  // 1024
  const int attn_blocks  = NB * NH * (NG / 16) / 8;         // 128

  for (int l = 0; l < NL; ++l) {
    const _Float16* wq16 = w16 + (size_t)l * ND * ND;
    const _Float16* wk16 = w16 + WMAT + (size_t)l * ND * ND;
    const _Float16* wv16 = w16 + 2 * WMAT + (size_t)l * ND * ND;
    const _Float16* wo16 = w16 + 3 * WMAT + (size_t)l * ND * ND;

    gemm_wmma<0><<<gemmG_blocks, 256, 0, stream>>>(
        hid16, wq16, bq + l * ND, dq, Q16, nullptr, NG, ND, ND, NB);
    gemm_wmma<0><<<gemmS_blocks, 256, 0, stream>>>(
        snpF, wk16, bk + l * ND, dkv, K16, nullptr, NS, ND, ND, NB);
    gemm_wmma<0><<<gemmS_blocks, 256, 0, stream>>>(
        snpF, wv16, bv + l * ND, nullptr, V16, nullptr, NS, ND, ND, NB);

    attn_flash<<<attn_blocks, 256, 0, stream>>>(Q16, K16, V16, cmask, att16);

    gemm_wmma<1><<<gemmG_blocks, 256, 0, stream>>>(
        att16, wo16, bo + l * ND, nullptr, nullptr, tmp32, NG, ND, ND, NB);

    ln_residual<<<(NB * NG) / 8, 256, 0, stream>>>(tmp32, lng + l * ND, lnb + l * ND, hid32, hid16);
  }

  film_head<<<(NB * NG) / 8, 256, 0, stream>>>(
      hid32, gam, bet, hw1, hb1, hlg, hlb, hw2, hb2, out);
}
